// TF_CAM_52793738002612
// MI455X (gfx1250) — matrix operand order repaired and checked by reading, not compiled, not run on hardware
//
#include <hip/hip_runtime.h>
#include <cstdint>

// ---------------------------------------------------------------------------
// NetVLAD-style fused pipeline for MI455X (gfx1250, wave32, WMMA, TDM).
// Kernel 1: per-image streaming pass: TDM async x-tile -> LDS, channel L2
//           norm -> bf16, GEMM1 (att = W*xn) via v_wmma_f32_16x16x32_bf16,
//           softmax over clusters, GEMM2 (out += att*xn^T) accumulated in
//           registers. TDM prefetch of next tile overlaps the GEMMs.
// Kernel 2: per-image normalization epilogue (row L2 + global L2) in LDS.
// ---------------------------------------------------------------------------

typedef __attribute__((ext_vector_type(16))) __bf16 bf16x16;
typedef __attribute__((ext_vector_type(8)))  __bf16 bf16x8;
typedef __attribute__((ext_vector_type(8)))  float  f32x8;
typedef __attribute__((ext_vector_type(4)))  unsigned int u32x4;
typedef __attribute__((ext_vector_type(8)))  int    i32x8;
typedef __attribute__((ext_vector_type(4)))  int    i32x4;

#define DIM_D   768
#define DIM_L   576
#define DIM_K   64
#define LT      32           // pixels per tile
#define NTILES  18           // 576 / 32
#define KSTEPS  24           // 768 / 32

// LDS element strides (chosen so every wmma fragment load is 16B aligned
// and scatter writes spread across banks)
#define W_STRIDE    776      // bf16 elements per W row   (768 + 8)
#define XNT_STRIDE  776      // xn as [l][d] (GEMM1 B)
#define XNN_STRIDE  40       // xn as [d][l] (GEMM2 B)    (32 + 8)
#define ATTF_STRIDE 33       // f32 logits [k][l]
#define ATTB_STRIDE 40       // bf16 softmaxed att [k][l] (GEMM2 A)

#define OFF_W     0
#define OFF_RAW   (OFF_W    + DIM_K * W_STRIDE   * 2)   //  99328
#define OFF_XNT   (OFF_RAW  + DIM_D * LT         * 4)   // +98304
#define OFF_XNN   (OFF_XNT  + LT    * XNT_STRIDE * 2)   // +49664
#define OFF_ATTF  (OFF_XNN  + DIM_D * XNN_STRIDE * 2)   // +61440
#define OFF_ATTB  (OFF_ATTF + DIM_K * ATTF_STRIDE* 4)   // + 8448
#define OFF_RED   (OFF_ATTB + DIM_K * ATTB_STRIDE* 2)   // + 5120
#define OFF_CSC   (OFF_RED  + 16 * 32 * 4)              // + 2048
#define SMEM1_SIZE (OFF_CSC + 32 * 4)                   // = 324480 <= 320KB

__device__ __forceinline__ unsigned short f2bf(float f) {
  unsigned int u = __float_as_uint(f);
  u += 0x7FFFu + ((u >> 16) & 1u);            // round-to-nearest-even
  return (unsigned short)(u >> 16);
}

__device__ __forceinline__ bf16x16 cat16(bf16x8 lo, bf16x8 hi) {
  return __builtin_shufflevector(lo, hi, 0, 1, 2, 3, 4, 5, 6, 7,
                                          8, 9, 10, 11, 12, 13, 14, 15);
}

__device__ __forceinline__ bf16x8 ld8(const unsigned short* p) {
  return *(const bf16x8*)p;
}

__launch_bounds__(512, 1)
__global__ void vlad_main(const float* __restrict__ x,
                          const float* __restrict__ w,
                          float* __restrict__ outraw) {
  extern __shared__ unsigned char smem[];
  unsigned short* Wl   = (unsigned short*)(smem + OFF_W);
  float*          raw  = (float*)         (smem + OFF_RAW);
  unsigned short* xnT  = (unsigned short*)(smem + OFF_XNT);
  unsigned short* xnN  = (unsigned short*)(smem + OFF_XNN);
  float*          attF = (float*)         (smem + OFF_ATTF);
  unsigned short* attB = (unsigned short*)(smem + OFF_ATTB);
  float*          red  = (float*)         (smem + OFF_RED);
  float*          csc  = (float*)         (smem + OFF_CSC);

  const int n    = blockIdx.x;
  const int t    = threadIdx.x;
  const int wave = t >> 5;
  const int lane = t & 31;
  const int half = lane >> 4;   // K-half for A/B fragments
  const int l16  = lane & 15;

  // --- stage conv_w as bf16 (once per image) ---
  for (int i = t; i < DIM_K * DIM_D; i += 512) {
    int k = i / DIM_D, d = i - k * DIM_D;
    Wl[k * W_STRIDE + d] = f2bf(w[i]);
  }

  // --- per-wave accumulators: krows 0..3  x  dcols {3w,3w+1,3w+2} ---
  f32x8 acc[4][3];
#pragma unroll
  for (int a = 0; a < 4; ++a)
#pragma unroll
    for (int b = 0; b < 3; ++b) acc[a][b] = f32x8{};

  // --- TDM: async 2-D tile load x[n][:, l0:l0+32] (f32) -> raw LDS ---
  auto issue_tdm = [&](int lt) {
    unsigned long long ga = (unsigned long long)(uintptr_t)(
        x + (size_t)n * DIM_D * DIM_L + (size_t)lt * LT);
    u32x4 g0;
    g0[0] = 1u;                                        // count=1, user D#
    g0[1] = (unsigned int)(uintptr_t)raw;              // lds_addr
    g0[2] = (unsigned int)ga;                          // global_addr lo
    g0[3] = ((unsigned int)(ga >> 32) & 0x01FFFFFFu)   // global_addr hi
            | 0x80000000u;                             // type=2 (image)
    i32x8 g1;
    g1[0] = (int)(2u << 16);            // data_size = 4B
    g1[1] = (int)(576u << 16);          // tensor_dim0[15:0]
    g1[2] = (int)(768u << 16);          // dim0 hi | tensor_dim1[15:0]
    g1[3] = (int)(32u << 16);           // dim1 hi | tile_dim0 = 32
    g1[4] = (int)768;                   // tile_dim1 = 768, tile_dim2 = 0
    g1[5] = (int)576;                   // tensor_dim0_stride = 576
    g1[6] = 0;
    g1[7] = 0;
    i32x4 gz4 = {0, 0, 0, 0};
    i32x8 gz8 = {0, 0, 0, 0, 0, 0, 0, 0};
    __builtin_amdgcn_tensor_load_to_lds(g0, g1, gz4, gz4, gz8, 0);
  };

  if (wave == 0) issue_tdm(0);
  __syncthreads();   // W staged; TDM issued

  for (int lt = 0; lt < NTILES; ++lt) {
    if (wave == 0) __builtin_amdgcn_s_wait_tensorcnt(0);
    __syncthreads();                     // raw tile visible to all waves

    // ---- channel-dim L2 normalize the 768x32 tile, emit bf16 twice ----
    const int lc = lane;                 // pixel column 0..31
    const int j  = wave;                 // 16 partial-sum groups
    float s = 0.f;
#pragma unroll 8
    for (int i = 0; i < 48; ++i) {
      float v = raw[(j + 16 * i) * LT + lc];
      s += v * v;
    }
    red[j * 32 + lc] = s;
    __syncthreads();
    if (j == 0) {
      float tot = 0.f;
#pragma unroll
      for (int jj = 0; jj < 16; ++jj) tot += red[jj * 32 + lc];
      csc[lc] = 1.f / fmaxf(sqrtf(tot), 1e-12f);
    }
    __syncthreads();
    {
      const float sc = csc[lc];
#pragma unroll 8
      for (int i = 0; i < 48; ++i) {
        const int d = j + 16 * i;
        const unsigned short b = f2bf(raw[d * LT + lc] * sc);
        xnT[lc * XNT_STRIDE + d] = b;    // [l][d]: GEMM1 B (K=d contiguous)
        xnN[d * XNN_STRIDE + lc] = b;    // [d][l]: GEMM2 B (K=l contiguous)
      }
    }
    __syncthreads();                     // raw now free -> prefetch next tile

    if (wave == 0 && lt + 1 < NTILES) issue_tdm(lt + 1);

    // ---- GEMM1: att(64 x 32) = W(64x768) * xn(768x32), waves 0..7 ----
    if (wave < 8) {
      const int kr   = wave >> 1;              // cluster-row tile 0..3
      const int lcol = (wave & 1) * 16 + l16;  // pixel column 0..31
      const int m    = kr * 16 + l16;          // A row
      f32x8 c = f32x8{};
      for (int kk = 0; kk < KSTEPS; ++kk) {
        const int k0 = kk * 32;
        // A: lanes 0-15 K={0..7,16..23}; lanes 16-31 K={8..15,24..31}
        bf16x16 a = cat16(ld8(Wl + m * W_STRIDE + k0 + 8 * half),
                          ld8(Wl + m * W_STRIDE + k0 + 16 + 8 * half));
        // B: lanes 0-15 K={0..7,16..23}; lanes 16-31 K={8..15,24..31}
        bf16x16 b = cat16(ld8(xnT + lcol * XNT_STRIDE + k0 + 8 * half),
                          ld8(xnT + lcol * XNT_STRIDE + k0 + 16 + 8 * half));
        c = __builtin_amdgcn_wmma_f32_16x16x32_bf16(
                false, a, false, b, (short)0, c, false, false);
      }
#pragma unroll
      for (int r = 0; r < 8; ++r)
        attF[(kr * 16 + 8 * half + r) * ATTF_STRIDE + lcol] = c[r];
    }
    __syncthreads();

    // ---- softmax over 64 clusters per pixel column ----
    if (t < 32) {
      float mx = -3.4e38f;
      for (int k = 0; k < DIM_K; ++k)
        mx = fmaxf(mx, attF[k * ATTF_STRIDE + t]);
      float sum = 0.f;
      for (int k = 0; k < DIM_K; ++k)
        sum += __expf(attF[k * ATTF_STRIDE + t] - mx);
      const float inv = 1.f / sum;
      for (int k = 0; k < DIM_K; ++k)
        attB[k * ATTB_STRIDE + t] =
            f2bf(__expf(attF[k * ATTF_STRIDE + t] - mx) * inv);
    }
    __syncthreads();

    // ---- GEMM2: out(64x768) += att(64x32) * xn^T(32x768), all 16 waves ----
#pragma unroll
    for (int kr = 0; kr < 4; ++kr) {
      const int krow = kr * 16 + l16;
      bf16x16 a = cat16(ld8(attB + krow * ATTB_STRIDE + 8 * half),
                        ld8(attB + krow * ATTB_STRIDE + 16 + 8 * half));
#pragma unroll
      for (int i = 0; i < 3; ++i) {
        const int d = (wave * 3 + i) * 16 + l16;
        bf16x16 b = cat16(ld8(xnN + d * XNN_STRIDE + 8 * half),
                          ld8(xnN + d * XNN_STRIDE + 16 + 8 * half));
        acc[kr][i] = __builtin_amdgcn_wmma_f32_16x16x32_bf16(
            false, a, false, b, (short)0, acc[kr][i], false, false);
      }
    }
    __syncthreads();   // protect xnT/xnN/attF before next tile
  }

  // ---- store un-normalized out[n] (f32) ----
  const size_t base = (size_t)n * (DIM_K * DIM_D);
#pragma unroll
  for (int kr = 0; kr < 4; ++kr)
#pragma unroll
    for (int i = 0; i < 3; ++i) {
      const int col = (wave * 3 + i) * 16 + l16;
#pragma unroll
      for (int r = 0; r < 8; ++r) {
        const int row = kr * 16 + 8 * half + r;
        outraw[base + (size_t)row * DIM_D + col] = acc[kr][i][r];
      }
    }
}

// --------------------------- epilogue kernel ------------------------------
#define SM2_BUF  0
#define SM2_RED  (49152 * 4)
#define SM2_RS   (SM2_RED + 64 * 4 * 4)
#define SM2_SR   (SM2_RS + 64 * 4)
#define SM2_G    (SM2_SR + 64 * 4)
#define SMEM2_SIZE (SM2_G + 16)

__launch_bounds__(256, 1)
__global__ void vlad_norm(float* __restrict__ out) {
  extern __shared__ unsigned char smem[];
  float* buf = (float*)(smem + SM2_BUF);   // [64][768] raw vlad for image n
  float* red = (float*)(smem + SM2_RED);   // [64][4] partial sumsq
  float* rs  = (float*)(smem + SM2_RS);    // [64] 1/row-norm
  float* sr  = (float*)(smem + SM2_SR);    // [64] row sumsq
  float* gsp = (float*)(smem + SM2_G);     // [1] 1/global-norm

  const int n = blockIdx.x;
  const int t = threadIdx.x;
  const size_t base = (size_t)n * 49152;

  for (int i = t; i < 49152; i += 256) buf[i] = out[base + i];
  __syncthreads();

  const int k = t >> 2, p = t & 3;         // 4 threads per cluster row
  float s = 0.f;
  for (int i = 0; i < 192; ++i) {
    float v = buf[k * DIM_D + p + 4 * i];
    s += v * v;
  }
  red[t] = s;
  __syncthreads();
  if (p == 0) {
    float tot = red[t] + red[t + 1] + red[t + 2] + red[t + 3];
    sr[k] = tot;
    rs[k] = 1.f / fmaxf(sqrtf(tot), 1e-12f);
  }
  __syncthreads();
  if (t == 0) {
    float g = 0.f;
    for (int kk = 0; kk < DIM_K; ++kk) g += sr[kk] * rs[kk] * rs[kk];
    gsp[0] = 1.f / fmaxf(sqrtf(g), 1e-12f);
  }
  __syncthreads();
  const float g = gsp[0];
  for (int i = t; i < 49152; i += 256)
    out[base + i] = buf[i] * rs[i / DIM_D] * g;
}

// ---------------------------------------------------------------------------
extern "C" void kernel_launch(void* const* d_in, const int* in_sizes, int n_in,
                              void* d_out, int out_size, void* d_ws,
                              size_t ws_size, hipStream_t stream) {
  (void)in_sizes; (void)n_in; (void)out_size; (void)d_ws; (void)ws_size;
  const float* x = (const float*)d_in[0];   // [64,768,24,24] f32
  const float* w = (const float*)d_in[1];   // [64,768] f32
  float* out = (float*)d_out;               // [64, 49152] f32

  vlad_main<<<64, 512, SMEM1_SIZE, stream>>>(x, w, out);
  vlad_norm<<<64, 256, SMEM2_SIZE, stream>>>(out);
}
